// QWenAttention_15710990369699
// MI455X (gfx1250) — compile-verified, matrix operand order checked by reading, not run on hardware
//
#include <hip/hip_runtime.h>
#include <hip/hip_bf16.h>
#include <stdint.h>

// ---------------------------------------------------------------------------
// QWen attention layer for MI455X (gfx1250, wave32, WMMA + async-LDS copies).
// B=2, S=2048, H=4096, 32 heads, head_dim=128.
//
// Pipeline:
//   0. convert hidden fp32 -> f16 (A16, aliases O16 region: disjoint lifetime)
//   1. transpose+convert w_qkv -> Wqkvt f16 [12288][4096], w_proj -> Wprojt
//   2. QKV GEMM: WMMA f16, double-buffered LDS staged with
//      global_load_async_to_lds_b128 (ASYNCcnt), epilogue adds bias and
//      scatters Q16/K16 [bh][s][d] and Vt16 [bh][d][s]
//   3. NeoX RoPE in-place on Q16/K16
//   4. Flash attention (one wave per 16-row Q tile), QK^T and P*V via
//      v_wmma_f32_16x16x32_f16, online softmax with shfl_xor reductions
//   5. Output projection GEMM (same async-staged mainloop) -> d_out fp32
//
// Workspace layout (268,435,456 bytes total):
//   [0)          Wqkvt  12288*4096 f16 = 100663296 B
//   [100663296)  Wprojt  4096*4096 f16 =  33554432 B
//   [134217728)  Q16    64*2048*128 f16 = 33554432 B
//   [167772160)  K16                     33554432 B
//   [201326592)  Vt16                    33554432 B
//   [234881024)  O16 / A16 (aliased)     33554432 B
// ---------------------------------------------------------------------------

typedef __attribute__((ext_vector_type(16))) _Float16 v16h;
typedef __attribute__((ext_vector_type(8)))  _Float16 v8h;
typedef __attribute__((ext_vector_type(8)))  float    v8f;
typedef __attribute__((ext_vector_type(4)))  float    v4f;

#define S_LEN 2048
#define H_DIM 4096
#define NH    32
#define HD    128
#define NQKV  12288
#define TILE_HALVES (128 * 32)

static __device__ __forceinline__ v16h make_frag(const _Float16* lo, const _Float16* hi) {
  v8h a = *(const v8h*)lo;
  v8h b = *(const v8h*)hi;
  v16h f;
#pragma unroll
  for (int i = 0; i < 8; ++i) { f[i] = a[i]; f[8 + i] = b[i]; }
  return f;
}

static __device__ __forceinline__ v8f wmma_f16(v16h a, v16h b, v8f c) {
  // D = A(16x32 f16) x B(32x16 f16) + C(16x16 f32)
  return __builtin_amdgcn_wmma_f32_16x16x32_f16(false, a, false, b, (short)0, c,
                                                false, false);
}

// Async global->LDS copy (tracked by ASYNCcnt, bypasses VGPRs).
static __device__ __forceinline__ void async_b128(const _Float16* g, _Float16* l) {
  const unsigned lds_off = (unsigned)(uintptr_t)l;   // addr[31:0] == LDS offset
  asm volatile("global_load_async_to_lds_b128 %0, %1, off"
               :: "v"(lds_off), "v"((unsigned long long)(uintptr_t)g)
               : "memory");
}

static __device__ __forceinline__ void wait_async0() {
  asm volatile("s_wait_asynccnt 0" ::: "memory");
}

// Stage one 128x32 f16 tile into LDS: 512 chunks of 8 halves, 2 per thread.
static __device__ __forceinline__ void async_load_tile(
    const _Float16* __restrict__ gbase, int rowBase, int k0,
    _Float16* lds, int tid) {
#pragma unroll
  for (int i = 0; i < 2; ++i) {
    const int c = tid + i * 256;
    const int row = c >> 2, kc = (c & 3) * 8;
    async_b128(gbase + (size_t)(rowBase + row) * H_DIM + k0 + kc,
               lds + row * 32 + kc);
  }
}

// Double-buffered WMMA mainloop: block=128x128 tile, 8 waves, wave=32x64.
static __device__ __forceinline__ void gemm_mainloop(
    const _Float16* __restrict__ A, const _Float16* __restrict__ Bm,
    int mBase, int nBase, _Float16* As, _Float16* Bs, int tid, v8f acc[2][4]) {
  const int lane = tid & 31, kh = lane >> 4, ln = lane & 15;
  const int wv = tid >> 5, wr = wv >> 1, wc = wv & 1;

#pragma unroll
  for (int mi = 0; mi < 2; ++mi)
#pragma unroll
    for (int ni = 0; ni < 4; ++ni)
#pragma unroll
      for (int r = 0; r < 8; ++r) acc[mi][ni][r] = 0.0f;

  // prologue: fill buffer 0
  async_load_tile(A, mBase, 0, As, tid);
  async_load_tile(Bm, nBase, 0, Bs, tid);
  wait_async0();
  __syncthreads();

  for (int k0 = 0; k0 < H_DIM; k0 += 32) {
    const int cur = (k0 >> 5) & 1;
    _Float16* Ac = As + cur * TILE_HALVES;
    _Float16* Bc = Bs + cur * TILE_HALVES;
    if (k0 + 32 < H_DIM) {           // kick off next tile while computing
      async_load_tile(A, mBase, k0 + 32, As + (cur ^ 1) * TILE_HALVES, tid);
      async_load_tile(Bm, nBase, k0 + 32, Bs + (cur ^ 1) * TILE_HALVES, tid);
    }
    v16h af[2], bf[4];
#pragma unroll
    for (int mi = 0; mi < 2; ++mi) {
      const int row = wr * 32 + mi * 16 + ln;
      af[mi] = make_frag(&Ac[row * 32 + kh * 8], &Ac[row * 32 + 16 + kh * 8]);
    }
#pragma unroll
    for (int ni = 0; ni < 4; ++ni) {
      const int col = wc * 64 + ni * 16 + ln;
      bf[ni] = make_frag(&Bc[col * 32 + kh * 8], &Bc[col * 32 + 16 + kh * 8]);
    }
#pragma unroll
    for (int mi = 0; mi < 2; ++mi)
#pragma unroll
      for (int ni = 0; ni < 4; ++ni)
        acc[mi][ni] = wmma_f16(af[mi], bf[ni], acc[mi][ni]);
    wait_async0();       // own async loads for the next buffer complete
    __syncthreads();     // everyone done reading cur + everyone's loads landed
  }
}

// ---------------------------------------------------------------------------
// 0. fp32 -> f16 elementwise convert (8 elements/thread)
// ---------------------------------------------------------------------------
__global__ void __launch_bounds__(256) k_cvt_f16(
    const float* __restrict__ in, _Float16* __restrict__ out) {
  const size_t i = ((size_t)blockIdx.x * 256 + threadIdx.x) * 8;
  v4f f0 = *(const v4f*)(in + i);
  v4f f1 = *(const v4f*)(in + i + 4);
  v8h h;
#pragma unroll
  for (int j = 0; j < 4; ++j) { h[j] = (_Float16)f0[j]; h[4 + j] = (_Float16)f1[j]; }
  *(v8h*)(out + i) = h;
}

// ---------------------------------------------------------------------------
// 1. Transpose + fp32->f16 convert:  in [K][N] fp32  ->  out [N][K] f16
// ---------------------------------------------------------------------------
__global__ void __launch_bounds__(256) k_transpose_cvt(
    const float* __restrict__ in, _Float16* __restrict__ out, int K, int N) {
  __shared__ float tile[32][33];
  const int tx = threadIdx.x, ty = threadIdx.y;
  const int x = blockIdx.x * 32 + tx;      // N index
  const int ybase = blockIdx.y * 32;       // K base
#pragma unroll
  for (int j = 0; j < 4; ++j)
    tile[ty + 8 * j][tx] = in[(size_t)(ybase + ty + 8 * j) * N + x];
  __syncthreads();
  const int yo = blockIdx.x * 32;          // N base in output
  const int xo = ybase + tx;               // K index in output row
#pragma unroll
  for (int j = 0; j < 4; ++j)
    out[(size_t)(yo + ty + 8 * j) * K + xo] = (_Float16)tile[tx][ty + 8 * j];
}

// ---------------------------------------------------------------------------
// 2. QKV GEMM: [4096][12288] = A16 x Wqkvt^T + bias, scatter epilogue
// ---------------------------------------------------------------------------
__global__ void __launch_bounds__(256) k_gemm_qkv(
    const _Float16* __restrict__ A, const _Float16* __restrict__ Wt,
    const float* __restrict__ bias, _Float16* __restrict__ Q16,
    _Float16* __restrict__ K16, _Float16* __restrict__ Vt16) {
  __shared__ _Float16 As[2 * TILE_HALVES];
  __shared__ _Float16 Bs[2 * TILE_HALVES];
  const int tid = threadIdx.x;
  const int lane = tid & 31, kh = lane >> 4, ln = lane & 15;
  const int wv = tid >> 5, wr = wv >> 1, wc = wv & 1;
  const int mBase = blockIdx.y * 128;
  const int nBase = blockIdx.x * 128;

  v8f acc[2][4];
  gemm_mainloop(A, Wt, mBase, nBase, As, Bs, tid, acc);

#pragma unroll
  for (int mi = 0; mi < 2; ++mi)
#pragma unroll
    for (int ni = 0; ni < 4; ++ni) {
      const int N = nBase + wc * 64 + ni * 16 + ln;
      const float bv = bias[N];
      const int which = N >> 12;            // 0=q 1=k 2=v
      const int rem = N & 4095;
      const int head = rem >> 7;
      const int d = rem & 127;
#pragma unroll
      for (int r = 0; r < 8; ++r) {
        const int M = mBase + wr * 32 + mi * 16 + r + 8 * kh;
        const int b = M >> 11, s = M & 2047;
        const float val = acc[mi][ni][r] + bv;
        const size_t bh = (size_t)b * NH + head;
        if (which == 0)
          Q16[(bh * S_LEN + s) * HD + d] = (_Float16)val;
        else if (which == 1)
          K16[(bh * S_LEN + s) * HD + d] = (_Float16)val;
        else
          Vt16[(bh * HD + d) * S_LEN + s] = (_Float16)val;
      }
    }
}

// ---------------------------------------------------------------------------
// 3. NeoX RoPE in-place on Q16/K16. One thread per (bh, s, pair d<64).
// ---------------------------------------------------------------------------
__global__ void __launch_bounds__(256) k_rope(
    const int* __restrict__ pos, _Float16* __restrict__ Q, _Float16* __restrict__ K) {
  const int idx = blockIdx.x * 256 + threadIdx.x;  // [0, 64*2048*64)
  const int d = idx & 63;
  const int s = (idx >> 6) & (S_LEN - 1);
  const int bh = idx >> 17;
  const int b = bh >> 5;
  const float p = (float)pos[b * S_LEN + s];
  // inv_freq = theta^(-2d/128); ln(10000) = 9.210340371976184
  const float freq = __expf(-(float)(2 * d) * (1.0f / 128.0f) * 9.210340371976184f);
  const float ang = p * freq;
  const float c = __cosf(ang), sn = __sinf(ang);
  const size_t base = ((size_t)bh * S_LEN + s) * HD + d;
  {
    const float x1 = (float)Q[base], x2 = (float)Q[base + 64];
    Q[base] = (_Float16)(x1 * c - x2 * sn);
    Q[base + 64] = (_Float16)(x2 * c + x1 * sn);
  }
  {
    const float x1 = (float)K[base], x2 = (float)K[base + 64];
    K[base] = (_Float16)(x1 * c - x2 * sn);
    K[base + 64] = (_Float16)(x2 * c + x1 * sn);
  }
}

// ---------------------------------------------------------------------------
// 4. Flash attention. Grid = (S/16 q-tiles, B*NH heads), one wave per block.
// ---------------------------------------------------------------------------
__global__ void __launch_bounds__(32) k_attn(
    const _Float16* __restrict__ Q, const _Float16* __restrict__ Kg,
    const _Float16* __restrict__ Vt, _Float16* __restrict__ O) {
  __shared__ _Float16 pbuf[16 * 32];
  const int lane = threadIdx.x;
  const int kh = lane >> 4, ln = lane & 15;
  const int bh = blockIdx.y;
  const int b = bh >> 5, h = bh & 31;
  const size_t hoff = (size_t)bh * S_LEN * HD;     // same size for Vt ([HD][S])
  const _Float16* Qp = Q + hoff;
  const _Float16* Kp = Kg + hoff;
  const _Float16* Vp = Vt + hoff;
  const int qbase = blockIdx.x * 16;
  const int qrow = qbase + ln;

  v16h aq[4];
#pragma unroll
  for (int dc = 0; dc < 4; ++dc) {
    const _Float16* p = Qp + (size_t)qrow * HD + dc * 32 + kh * 8;
    aq[dc] = make_frag(p, p + 16);
  }

  float mrow[8], lrow[8];
  v8f oacc[8];
#pragma unroll
  for (int r = 0; r < 8; ++r) { mrow[r] = -3.0e38f; lrow[r] = 0.0f; }
#pragma unroll
  for (int dc = 0; dc < 8; ++dc)
#pragma unroll
    for (int r = 0; r < 8; ++r) oacc[dc][r] = 0.0f;

  const float scale = 0.08838834764831845f;        // 1/sqrt(128)
  const int nkb = (qbase + 16 + 31) >> 5;          // causal: 32-key blocks

  for (int kb = 0; kb < nkb; ++kb) {
    const int k0 = kb * 32;
    v8f sacc[2];
#pragma unroll
    for (int t = 0; t < 2; ++t)
#pragma unroll
      for (int r = 0; r < 8; ++r) sacc[t][r] = 0.0f;

#pragma unroll
    for (int t = 0; t < 2; ++t) {
      const _Float16* kp = Kp + (size_t)(k0 + t * 16 + ln) * HD;
#pragma unroll
      for (int dc = 0; dc < 4; ++dc) {
        v16h bk = make_frag(kp + dc * 32 + kh * 8, kp + dc * 32 + 16 + kh * 8);
        sacc[t] = wmma_f16(aq[dc], bk, sacc[t]);
      }
    }

    // online softmax; rows live in 16-lane groups (lane>>4 selects row half)
#pragma unroll
    for (int r = 0; r < 8; ++r) {
      const int row = qbase + r + 8 * kh;
      float x0 = sacc[0][r] * scale;
      float x1 = sacc[1][r] * scale;
      if (k0 + ln > row) x0 = -3.0e38f;
      if (k0 + 16 + ln > row) x1 = -3.0e38f;
      float mx = fmaxf(x0, x1);
      mx = fmaxf(mx, __shfl_xor(mx, 1));
      mx = fmaxf(mx, __shfl_xor(mx, 2));
      mx = fmaxf(mx, __shfl_xor(mx, 4));
      mx = fmaxf(mx, __shfl_xor(mx, 8));
      const float mn = fmaxf(mrow[r], mx);
      const float corr = __expf(mrow[r] - mn);
      mrow[r] = mn;
      const float p0 = __expf(x0 - mn);
      const float p1 = __expf(x1 - mn);
      float rs = p0 + p1;
      rs += __shfl_xor(rs, 1);
      rs += __shfl_xor(rs, 2);
      rs += __shfl_xor(rs, 4);
      rs += __shfl_xor(rs, 8);
      lrow[r] = lrow[r] * corr + rs;
#pragma unroll
      for (int dc = 0; dc < 8; ++dc) oacc[dc][r] *= corr;
      pbuf[(r + 8 * kh) * 32 + ln] = (_Float16)p0;
      pbuf[(r + 8 * kh) * 32 + 16 + ln] = (_Float16)p1;
    }
    __syncthreads();

    v16h ap = make_frag(&pbuf[ln * 32 + kh * 8], &pbuf[ln * 32 + 16 + kh * 8]);
#pragma unroll
    for (int dc = 0; dc < 8; ++dc) {
      const _Float16* vp = Vp + (size_t)(dc * 16 + ln) * S_LEN + k0 + kh * 8;
      v16h bv = make_frag(vp, vp + 16);
      oacc[dc] = wmma_f16(ap, bv, oacc[dc]);
    }
    __syncthreads();
  }

  // normalize + store f16 row-major [token][H] for the projection GEMM
#pragma unroll
  for (int r = 0; r < 8; ++r) {
    const float inv = 1.0f / lrow[r];
    const size_t token = (size_t)b * S_LEN + (qbase + r + 8 * kh);
#pragma unroll
    for (int dc = 0; dc < 8; ++dc)
      O[token * H_DIM + h * HD + dc * 16 + ln] = (_Float16)(oacc[dc][r] * inv);
  }
}

// ---------------------------------------------------------------------------
// 5. Output projection: out(fp32)[4096][4096] = O16 x Wprojt^T
// ---------------------------------------------------------------------------
__global__ void __launch_bounds__(256) k_gemm_proj(
    const _Float16* __restrict__ A, const _Float16* __restrict__ Wt,
    float* __restrict__ out) {
  __shared__ _Float16 As[2 * TILE_HALVES];
  __shared__ _Float16 Bs[2 * TILE_HALVES];
  const int tid = threadIdx.x;
  const int lane = tid & 31, kh = lane >> 4, ln = lane & 15;
  const int wv = tid >> 5, wr = wv >> 1, wc = wv & 1;
  const int mBase = blockIdx.y * 128;
  const int nBase = blockIdx.x * 128;

  v8f acc[2][4];
  gemm_mainloop(A, Wt, mBase, nBase, As, Bs, tid, acc);

#pragma unroll
  for (int mi = 0; mi < 2; ++mi)
#pragma unroll
    for (int ni = 0; ni < 4; ++ni) {
      const int N = nBase + wc * 64 + ni * 16 + ln;
#pragma unroll
      for (int r = 0; r < 8; ++r) {
        const int M = mBase + wr * 32 + mi * 16 + r + 8 * kh;
        out[(size_t)M * H_DIM + N] = acc[mi][ni][r];
      }
    }
}

// ---------------------------------------------------------------------------
extern "C" void kernel_launch(void* const* d_in, const int* in_sizes, int n_in,
                              void* d_out, int out_size, void* d_ws, size_t ws_size,
                              hipStream_t stream) {
  const int*   positions = (const int*)d_in[0];
  const float* hidden    = (const float*)d_in[1];
  const float* w_qkv     = (const float*)d_in[2];
  const float* b_qkv     = (const float*)d_in[3];
  const float* w_proj    = (const float*)d_in[4];
  float* out = (float*)d_out;

  char* ws = (char*)d_ws;
  _Float16* Wqkvt  = (_Float16*)(ws + 0);
  _Float16* Wprojt = (_Float16*)(ws + 100663296);
  _Float16* Q16    = (_Float16*)(ws + 134217728);
  _Float16* K16    = (_Float16*)(ws + 167772160);
  _Float16* Vt16   = (_Float16*)(ws + 201326592);
  _Float16* O16    = (_Float16*)(ws + 234881024);
  _Float16* A16    = O16;   // aliased: hidden-f16 is dead before O16 is written
  // total workspace: 268435456 bytes

  // 0. hidden fp32 -> f16 (4096*4096 elements, 8 per thread)
  k_cvt_f16<<<(4096 * 4096) / (256 * 8), 256, 0, stream>>>(hidden, A16);

  // 1. weight transpose+convert (f16 w_qkv fits in 192MB L2 -> re-reads hit L2)
  k_transpose_cvt<<<dim3(NQKV / 32, H_DIM / 32), dim3(32, 8), 0, stream>>>(
      w_qkv, Wqkvt, H_DIM, NQKV);
  k_transpose_cvt<<<dim3(H_DIM / 32, H_DIM / 32), dim3(32, 8), 0, stream>>>(
      w_proj, Wprojt, H_DIM, H_DIM);

  // 2. QKV projection GEMM (M=4096, N=12288, K=4096), async-LDS double buffer
  k_gemm_qkv<<<dim3(NQKV / 128, 4096 / 128), 256, 0, stream>>>(
      A16, Wqkvt, b_qkv, Q16, K16, Vt16);

  // 3. RoPE: 64 heads * 2048 seq * 64 pairs
  k_rope<<<8388608 / 256, 256, 0, stream>>>(positions, Q16, K16);

  // 4. flash attention: (128 q-tiles, 64 batch*head), one wave each
  k_attn<<<dim3(S_LEN / 16, 2 * NH), 32, 0, stream>>>(Q16, K16, Vt16, O16);

  // 5. output projection (M=4096, N=4096, K=4096) -> fp32 d_out
  k_gemm_proj<<<dim3(H_DIM / 128, 4096 / 128), 256, 0, stream>>>(O16, Wprojt, out);
}
